// ProbabilisticFastRCNNOutputLayers_87110526698156
// MI455X (gfx1250) — compile-verified, matrix operand order checked by reading, not run on hardware
//
#include <hip/hip_runtime.h>
#include <cstdint>
#include <math.h>

// ---------------------------------------------------------------------------
// CDNA5 / gfx1250 feature detection
// ---------------------------------------------------------------------------
#if defined(__has_builtin)
#  if __has_builtin(__builtin_amdgcn_global_load_async_to_lds_b128)
#    define HAVE_ASYNC_LDS 1
#  endif
#  if __has_builtin(__builtin_amdgcn_s_wait_asynccnt)
#    define HAVE_WAIT_ASYNC 1
#  endif
#endif
#ifndef HAVE_ASYNC_LDS
#define HAVE_ASYNC_LDS 0
#endif
#ifndef HAVE_WAIT_ASYNC
#define HAVE_WAIT_ASYNC 0
#endif

typedef __attribute__((ext_vector_type(16))) _Float16 v16h;
typedef __attribute__((ext_vector_type(8)))  _Float16 v8h;
typedef __attribute__((ext_vector_type(8)))  float    v8f;
// async-to-LDS builtin operand type (per hipcc diagnostic: vector_size(16) int)
typedef int v4i __attribute__((vector_size(16)));

#define DEV __device__ __forceinline__

namespace {
constexpr int   R_      = 50000;
constexpr int   K_      = 80;
constexpr int   TOPK_   = 100;
constexpr int   NSAMP_  = 100;
constexpr int   N_      = R_ * K_;            // 4,000,000 candidates
constexpr float SCORE_TH = 0.05f;
constexpr float NMS_TH   = 0.5f;
constexpr float IMGH = 800.f, IMGW = 1333.f;
constexpr float SCL  = 4.135166556742356f;    // log(1000/16)
constexpr int   TILE_   = 256;
constexpr int   NTILES_ = N_ / TILE_;         // 15625 (exact)
constexpr int   NBLK_   = 512;                // scan blocks (8 waves each)
constexpr int   MP_ = 112;                    // MC rows padded to 7*16
constexpr int   KP_ = 128;                    // MC cols padded to 4*32

// workspace layout (bytes, all 16B aligned)
constexpr size_t OFF_OBX  = 0;                                  // N*4 f32 (offset boxes)
constexpr size_t OFF_MSC  = OFF_OBX  + (size_t)N_ * 4 * 4;      // N f32 (masked scores)
constexpr size_t OFF_PV   = OFF_MSC  + (size_t)N_ * 4;          // 2*NBLK f32 partial max (ping-pong)
constexpr size_t OFF_PI   = OFF_PV   + (size_t)2 * NBLK_ * 4;   // 2*NBLK i32 partial idx
constexpr size_t OFF_MAXC = OFF_PI   + (size_t)2 * NBLK_ * 4;   // 1 f32 (padded to 16)
constexpr size_t OFF_KB   = OFF_MAXC + 16;                      // TOPK*4 f32 raw kept boxes
constexpr size_t OFF_KCOV = OFF_KB   + (size_t)TOPK_ * 4 * 4;   // TOPK*4 f32 kept covs
constexpr size_t OFF_KVAL = OFF_KCOV + (size_t)TOPK_ * 4 * 4;   // TOPK i32 valid flags
} // namespace

// argmax with first-index tie-break (matches jnp.argmax)
DEV bool better(float v1, int i1, float v2, int i2) {
  return (v1 > v2) || (v1 == v2 && i1 < i2);
}

DEV float iou44(float4 a, float4 b) {
  float a1 = (a.z - a.x) * (a.w - a.y);
  float a2 = (b.z - b.x) * (b.w - b.y);
  float lx = fmaxf(a.x, b.x), ly = fmaxf(a.y, b.y);
  float rx = fminf(a.z, b.z), ry = fminf(a.w, b.w);
  float w = fmaxf(rx - lx, 0.f), h = fmaxf(ry - ly, 0.f);
  float inter = w * h;
  return inter > 0.f ? inter / fmaxf(a1 + a2 - inter, 1e-12f) : 0.f;
}

// ---- async global->LDS staging helpers (CDNA5 ASYNCcnt path) --------------
DEV void stage_async(const float4* gp, float4* lp) {
#if HAVE_ASYNC_LDS
  // addrspace(1)/addrspace(3) v4i pointers via integer casts: a flat global
  // address == the global address; low 32 bits of a flat LDS address == the
  // LDS offset (aperture check uses addr[31:0]).
  auto g = (__attribute__((address_space(1))) v4i*)(uintptr_t)gp;
  auto l = (__attribute__((address_space(3))) v4i*)(unsigned)(uintptr_t)lp;
  __builtin_amdgcn_global_load_async_to_lds_b128(g, l, 0, 0);
#else
  *lp = *gp;
#endif
}

template <int Nw>
DEV void wait_async() {
#if HAVE_ASYNC_LDS
#  if HAVE_WAIT_ASYNC
  __builtin_amdgcn_s_wait_asynccnt(Nw);
#  else
  if constexpr (Nw == 0) asm volatile("s_wait_asynccnt 0" ::: "memory");
  else                   asm volatile("s_wait_asynccnt 1" ::: "memory");
#  endif
#endif
}

// ---------------------------------------------------------------------------
// Phase A: init maxc + sentinel partials (ping-pong buffer 1, read at iter 0)
// ---------------------------------------------------------------------------
__global__ void k_init(float* maxc, float* pV, int* pI) {
  int j = blockIdx.x * blockDim.x + threadIdx.x;
  if (j == 0) *maxc = 0.f;
  if (j < NBLK_) {
    pV[NBLK_ + j] = -__builtin_inff();
    pI[NBLK_ + j] = 0x7fffffff;   // matches no real index -> no exclusion at iter 0
  }
}

__global__ void __launch_bounds__(256) k_maxred(const float4* __restrict__ boxes,
                                                float* __restrict__ maxc) {
  float m = 0.f;
  for (int i = blockIdx.x * blockDim.x + threadIdx.x; i < N_; i += gridDim.x * blockDim.x) {
    float4 b = boxes[i];
    m = fmaxf(m, fminf(fmaxf(b.x, 0.f), IMGW));
    m = fmaxf(m, fminf(fmaxf(b.y, 0.f), IMGH));
    m = fmaxf(m, fminf(fmaxf(b.z, 0.f), IMGW));
    m = fmaxf(m, fminf(fmaxf(b.w, 0.f), IMGH));
  }
  __shared__ float sm[256];
  sm[threadIdx.x] = m;
  __syncthreads();
  for (int s = 128; s > 0; s >>= 1) {
    if (threadIdx.x < s) sm[threadIdx.x] = fmaxf(sm[threadIdx.x], sm[threadIdx.x + s]);
    __syncthreads();
  }
  if (threadIdx.x == 0) atomicMax((int*)maxc, __float_as_int(sm[0])); // coords >= 0
}

// ---------------------------------------------------------------------------
// Phase B: build class-offset boxes + masked scores
// ---------------------------------------------------------------------------
__global__ void __launch_bounds__(256) k_build(const float4* __restrict__ boxes,
                                               const float* __restrict__ scores,
                                               const float* __restrict__ maxc,
                                               float4* __restrict__ obx,
                                               float* __restrict__ msc) {
  int i = blockIdx.x * 256 + threadIdx.x; // grid == NTILES_, covers N_ exactly
  float4 b = boxes[i];
  float x1 = fminf(fmaxf(b.x, 0.f), IMGW);
  float y1 = fminf(fmaxf(b.y, 0.f), IMGH);
  float x2 = fminf(fmaxf(b.z, 0.f), IMGW);
  float y2 = fminf(fmaxf(b.w, 0.f), IMGH);
  int cls = i % K_;
  int row = i / K_;
  float off = (float)cls * (maxc[0] + 1.f);
  obx[i] = make_float4(x1 + off, y1 + off, x2 + off, y2 + off);
  float s = scores[(size_t)row * (K_ + 1) + cls];
  msc[i] = (s > SCORE_TH) ? s : -__builtin_inff();
}

// ---------------------------------------------------------------------------
// Phase C: one kernel per NMS step (101 launches total).
//  1) every block redundantly reduces the previous iteration's partials
//     (deterministic, first-index tie-break) -> same selection in all blocks
//  2) block 0 records kept entry (iter-1) and persists msc[best] = -inf
//  3) all blocks scan: suppress vs selected box, block argmax -> partials
// Partials are ping-pong buffered so 3) never races 1) of slower blocks.
// Box stream staged to LDS with double-buffered async copies (ASYNCcnt).
// ---------------------------------------------------------------------------
__global__ void __launch_bounds__(256) k_nms_step(const float4* __restrict__ obx,
                                                  float* __restrict__ msc,
                                                  const float* __restrict__ covs,
                                                  const float* __restrict__ maxc,
                                                  float* __restrict__ pV,
                                                  int* __restrict__ pI,
                                                  float* __restrict__ kb,
                                                  float* __restrict__ kcov,
                                                  int* __restrict__ kval,
                                                  float* __restrict__ out,
                                                  int iter) {
  __shared__ float4 buf[2][TILE_];
  __shared__ float  sv[256];
  __shared__ int    si[256];
  const int   t    = threadIdx.x;
  const float NINF = -__builtin_inff();

  // ---- 1) reduce previous partials (read buffer = (iter-1)&1) ----
  const int rd = ((iter + 1) & 1) * NBLK_;
  float bv = NINF;
  int   bi = 0x7fffffff;
  for (int j = t; j < NBLK_; j += 256) {
    float v = pV[rd + j];
    int   x = pI[rd + j];
    if (better(v, x, bv, bi)) { bv = v; bi = x; }
  }
  sv[t] = bv; si[t] = bi;
  __syncthreads();
  for (int k = 128; k > 0; k >>= 1) {
    if (t < k && better(sv[t + k], si[t + k], sv[t], si[t])) { sv[t] = sv[t + k]; si[t] = si[t + k]; }
    __syncthreads();
  }
  const float prevV = sv[0];
  const int   prevB = si[0];
  __syncthreads();   // sv/si are reused by phase 3

  const bool valid = (prevV > NINF);
  const int  bEff  = valid ? prevB : 0;   // jnp.argmax over all -inf returns 0
  float4 sel = make_float4(0.f, 0.f, 0.f, 0.f);
  if (iter > 0) {
    float4 ob  = obx[bEff];               // uniform address: L2 broadcast
    if (valid) sel = ob;
    // ---- 2) designated writer records kept entry (iter-1) ----
    if (blockIdx.x == 0 && t == 0) {
      int   slot = iter - 1;
      int   cls  = bEff % K_;
      float off  = (float)cls * (maxc[0] + 1.f);
      float x1 = ob.x - off, y1 = ob.y - off, x2 = ob.z - off, y2 = ob.w - off;
      // out: [0..399] boxes, [400..499] scores, [500..599] cls, [600..699] mc_iou, [700..799] valid
      out[slot * 4 + 0] = valid ? x1 : 0.f;
      out[slot * 4 + 1] = valid ? y1 : 0.f;
      out[slot * 4 + 2] = valid ? x2 : 0.f;
      out[slot * 4 + 3] = valid ? y2 : 0.f;
      out[400 + slot] = valid ? prevV : 0.f;
      out[500 + slot] = valid ? (float)cls : -1.f;
      out[700 + slot] = valid ? 1.f : 0.f;
      kb[slot * 4 + 0] = x1; kb[slot * 4 + 1] = y1;  // raw (un-zeroed) for MC phase
      kb[slot * 4 + 2] = x2; kb[slot * 4 + 3] = y2;
      kcov[slot * 4 + 0] = covs[(size_t)bEff * 4 + 0];
      kcov[slot * 4 + 1] = covs[(size_t)bEff * 4 + 1];
      kcov[slot * 4 + 2] = covs[(size_t)bEff * 4 + 2];
      kcov[slot * 4 + 3] = covs[(size_t)bEff * 4 + 3];
      kval[slot] = valid ? 1 : 0;
      msc[bEff] = NINF;  // persist for later iterations (masked locally this iter)
    }
  }
  if (iter == TOPK_) return;  // finalize-only launch

  // ---- 3) scan: suppress + block argmax into write buffer iter&1 ----
  bv = NINF;
  bi = 0x7fffffff;
  int tile = blockIdx.x;
  int cur  = 0;
  if (tile < NTILES_) stage_async(&obx[(size_t)tile * TILE_ + t], &buf[0][t]);
  while (tile < NTILES_) {
    int next = tile + gridDim.x;
    if (next < NTILES_) {
      stage_async(&obx[(size_t)next * TILE_ + t], &buf[cur ^ 1][t]);
      __builtin_prefetch(&msc[(size_t)next * TILE_ + t], 0, 1); // -> global_prefetch_b8
      wait_async<1>(); // oldest (current tile) done; async loads complete in order
    } else {
      wait_async<0>();
    }
    int i = tile * TILE_ + t;
    float4 b = buf[cur][t];
    float sc = msc[i];
    if (iter > 0 && i == bEff) sc = NINF;  // this step's exclusion (global write raced)
    float iou = iou44(sel, b);
    if (iou > NMS_TH) { sc = NINF; msc[i] = sc; }
    if (better(sc, i, bv, bi)) { bv = sc; bi = i; }
    tile = next;
    cur ^= 1;
  }

  sv[t] = bv; si[t] = bi;
  __syncthreads();
  for (int k = 128; k > 0; k >>= 1) {
    if (t < k && better(sv[t + k], si[t + k], sv[t], si[t])) { sv[t] = sv[t + k]; si[t] = si[t + k]; }
    __syncthreads();
  }
  if (t == 0) {
    const int wr = (iter & 1) * NBLK_;
    pV[wr + blockIdx.x] = sv[0];
    pI[wr + blockIdx.x] = si[0];
  }
}

// ---------------------------------------------------------------------------
// Phase D+E fused: Monte-Carlo IoU samples into LDS (112x128 f16), then the
// per-box mean over samples on the matrix pipe:
//   D = A(112x128 f16, iou samples) x B(128x16, all 1/NSAMP)
// via v_wmma_f32_16x16x32_f16 reading A-fragments from LDS (ds_load_b128).
// ---------------------------------------------------------------------------
__global__ void __launch_bounds__(256) k_mc(const float* __restrict__ noise,
                                            const float* __restrict__ kb,
                                            const float* __restrict__ kcov,
                                            const int* __restrict__ kval,
                                            float* __restrict__ out) {
  __shared__ _Float16 S[MP_ * KP_];   // 28 KB of the 320 KB/WGP LDS
  const int t = threadIdx.x;

  // ---- samples -> LDS (rows >= 100 and cols >= 100 are zero padding) ----
  for (int e = t; e < MP_ * KP_; e += 256) {
    int b   = e >> 7;        // / KP_
    int smp = e & (KP_ - 1); // % KP_
    float val = 0.f;
    if (b < TOPK_ && smp < NSAMP_) {
      const float* n = noise + ((size_t)b * NSAMP_ + smp) * 4;
      float s0 = n[0] * sqrtf(kcov[b * 4 + 0]);
      float s1 = n[1] * sqrtf(kcov[b * 4 + 1]);
      float s2 = n[2] * sqrtf(kcov[b * 4 + 2]);
      float s3 = n[3] * sqrtf(kcov[b * 4 + 3]);
      float x1 = kb[b * 4 + 0], y1 = kb[b * 4 + 1];
      float x2 = kb[b * 4 + 2], y2 = kb[b * 4 + 3];
      float w = x2 - x1, h = y2 - y1;
      float cx = x1 + 0.5f * w, cy = y1 + 0.5f * h;
      float dx = s0 * 0.1f, dy = s1 * 0.1f;
      float dw = fminf(s2 * 0.2f, SCL), dh = fminf(s3 * 0.2f, SCL);
      float pcx = dx * w + cx, pcy = dy * h + cy;
      float pw = __expf(dw) * w, ph = __expf(dh) * h;
      float4 pb = make_float4(pcx - 0.5f * pw, pcy - 0.5f * ph,
                              pcx + 0.5f * pw, pcy + 0.5f * ph);
      val = iou44(make_float4(x1, y1, x2, y2), pb);
    }
    S[e] = (_Float16)val;
  }
  __syncthreads();

  // ---- WMMA mean-reduction: 7 row tiles (one wave each), 4 K-chunks ----
  const int lane = t & 31;
  const int wv   = t >> 5;
  if (wv >= MP_ / 16) return;  // whole-wave uniform: EXEC all-ones in active waves
  const int m    = lane & 15;
  const int half = lane >> 4;
  const int row  = wv * 16 + m;

  v16h bm;
  #pragma unroll
  for (int e = 0; e < 16; ++e) bm[e] = (_Float16)(1.0f / (float)NSAMP_);

  v8f acc = {};
  const v8h* Sr = (const v8h*)(S + (size_t)row * KP_);
  #pragma unroll
  for (int c = 0; c < KP_ / 32; ++c) {
    // 16-bit A-matrix 16x32 lane layout: lanes 0-15 hold K 0-7 (VGPR0-3) and
    // 16-23 (VGPR4-7); lanes 16-31 hold K 8-15 and 24-31. Both 8-element runs
    // are contiguous 16B LDS loads (ds_load_b128).
    v8h lo = Sr[c * 4 + half];      // K = c*32 + half*8 + [0..7]
    v8h hi = Sr[c * 4 + 2 + half];  // K = c*32 + 16 + half*8 + [0..7]
    v16h a;
    #pragma unroll
    for (int e = 0; e < 8; ++e) { a[e] = lo[e]; a[8 + e] = hi[e]; }
    acc = __builtin_amdgcn_wmma_f32_16x16x32_f16(false, a, false, bm, (short)0, acc,
                                                 false, false);
  }
  // C/D layout: lanes 0-15 => N=lane, M=j (VGPR j); lanes 16-31 => M=8+j.
  // Column 0 lives in lane 0 (M 0..7) and lane 16 (M 8..15).
  if (lane == 0 || lane == 16) {
    int base = wv * 16 + (lane == 16 ? 8 : 0);
    #pragma unroll
    for (int j = 0; j < 8; ++j) {
      int i = base + j;
      if (i < TOPK_) out[600 + i] = kval[i] ? acc[j] : 0.f;
    }
  }
}

// ---------------------------------------------------------------------------
extern "C" void kernel_launch(void* const* d_in, const int* in_sizes, int n_in,
                              void* d_out, int out_size, void* d_ws, size_t ws_size,
                              hipStream_t stream) {
  (void)in_sizes; (void)n_in; (void)out_size; (void)ws_size;
  const float* scores = (const float*)d_in[0];   // (R, K+1)
  const float* boxes  = (const float*)d_in[1];   // (R, K*4)
  const float* covs   = (const float*)d_in[2];   // (R, K*4)
  const float* noise  = (const float*)d_in[3];   // (TOPK, NSAMP, 4)
  float* out = (float*)d_out;

  char* ws = (char*)d_ws;
  float4* obx  = (float4*)(ws + OFF_OBX);
  float*  msc  = (float*)(ws + OFF_MSC);
  float*  pV   = (float*)(ws + OFF_PV);
  int*    pI   = (int*)(ws + OFF_PI);
  float*  maxc = (float*)(ws + OFF_MAXC);
  float*  kb   = (float*)(ws + OFF_KB);
  float*  kcov = (float*)(ws + OFF_KCOV);
  int*    kval = (int*)(ws + OFF_KVAL);

  k_init<<<(NBLK_ + 255) / 256, 256, 0, stream>>>(maxc, pV, pI);
  k_maxred<<<2048, 256, 0, stream>>>((const float4*)boxes, maxc);
  k_build<<<NTILES_, 256, 0, stream>>>((const float4*)boxes, scores, maxc, obx, msc);

  for (int it = 0; it <= TOPK_; ++it) {
    k_nms_step<<<NBLK_, 256, 0, stream>>>(obx, msc, covs, maxc, pV, pI, kb, kcov, kval,
                                          out, it);
  }

  k_mc<<<1, 256, 0, stream>>>(noise, kb, kcov, kval, out);
}
